// FusedAttention_63505386438868
// MI455X (gfx1250) — matrix operand order repaired and checked by reading, not compiled, not run on hardware
//
#include <hip/hip_runtime.h>
#include <stdint.h>

// ---------------------------------------------------------------------------
// Fused attention block for MI455X (gfx1250, wave32).
// All matmuls run on v_wmma_f32_16x16x32_bf16 (f32 accumulate).
// Weights are pre-converted to bf16 once; the GEMM hot loop stages them with
// GLOBAL_LOAD_ASYNC_TO_LDS_B128 (ASYNCcnt-tracked, no VGPR transit).
// Activations are converted on the fly with v_cvt_pk_bf16_f32.
// Workspace layout:
//   f32 : Q[2048*4096] K[2048*1024] V[2048*1024] A[2048*4096]   (83.9 MB)
//   bf16: wq[4096*4096] wk[1024*4096] wv[1024*4096] wo[4096*4096] (83.9 MB)
// ---------------------------------------------------------------------------

typedef __attribute__((ext_vector_type(16))) __bf16 v16bf;
typedef __attribute__((ext_vector_type(8)))  float  v8f;
typedef int v4i_t __attribute__((__vector_size__(4 * sizeof(int))));

union Frag { v16bf v; uint32_t u[8]; };
union Acc  { v8f  v; float   f[8]; };

struct alignas(16) U4 { uint32_t x, y, z, w; };
struct alignas(8)  U2 { uint32_t x, y; };

#if __has_builtin(__builtin_amdgcn_global_load_async_to_lds_b128) && \
    __has_builtin(__builtin_amdgcn_s_wait_asynccnt)
#define HAVE_ASYNC_LDS 1
#else
#define HAVE_ASYNC_LDS 0
#endif

__device__ __forceinline__ void async_cp_b128(const unsigned short* g,
                                              unsigned short* l) {
#if HAVE_ASYNC_LDS
  __builtin_amdgcn_global_load_async_to_lds_b128(
      (__attribute__((address_space(1))) v4i_t*)g,
      (__attribute__((address_space(3))) v4i_t*)l, 0, 0);
#else
  (void)g; (void)l;
#endif
}
__device__ __forceinline__ void wait_async0() {
#if HAVE_ASYNC_LDS
  __builtin_amdgcn_s_wait_asynccnt(0);
#endif
}

__device__ __forceinline__ unsigned short f2bf(float x) {
  union { __bf16 h; unsigned short s; } c;
  c.h = (__bf16)x;                       // lowers to native bf16 cvt
  return c.s;
}
__device__ __forceinline__ uint32_t pk2bf(float lo, float hi) {
#if __has_builtin(__builtin_amdgcn_cvt_pk_bf16_f32)
  auto r = __builtin_amdgcn_cvt_pk_bf16_f32(lo, hi);
  uint32_t u;
  __builtin_memcpy(&u, &r, 4);
  return u;
#else
  typedef __attribute__((ext_vector_type(2))) __bf16 v2bf;
  v2bf t;
  t.x = (__bf16)lo;
  t.y = (__bf16)hi;                      // matched to v_cvt_pk_bf16_f32
  uint32_t u;
  __builtin_memcpy(&u, &t, 4);
  return u;
#endif
}

__device__ __forceinline__ v8f wmma_bf16(v16bf a, v16bf b, v8f c) {
  return __builtin_amdgcn_wmma_f32_16x16x32_bf16(false, a, false, b,
                                                 (short)0, c, false, false);
}

// load 4 consecutive dwords (8 bf16) from 16B-aligned LDS into frag half
__device__ __forceinline__ void ld_half(Frag& f, int base, const unsigned short* p) {
  const U4 t = *(const U4*)p;
  f.u[base + 0] = t.x; f.u[base + 1] = t.y;
  f.u[base + 2] = t.z; f.u[base + 3] = t.w;
}

// ---------------------------------------------------------------------------
// One-time fp32 -> bf16 conversion (8 elements / thread, b128 in & out)
// ---------------------------------------------------------------------------
__global__ __launch_bounds__(256) void cvt_f32_bf16(
    const float* __restrict__ in, unsigned short* __restrict__ out, int n8)
{
  const int i = blockIdx.x * 256 + threadIdx.x;
  if (i >= n8) return;
  const float4 f0 = *(const float4*)(in + (size_t)i * 8);
  const float4 f1 = *(const float4*)(in + (size_t)i * 8 + 4);
  U4 r;
  r.x = pk2bf(f0.x, f0.y); r.y = pk2bf(f0.z, f0.w);
  r.z = pk2bf(f1.x, f1.y); r.w = pk2bf(f1.z, f1.w);
  *(U4*)(out + (size_t)i * 8) = r;
}

// ---------------------------------------------------------------------------
// C[M,N] = A[M,K](f32) * B[N,K](bf16)^T, f32 out.
// Block: 256 threads = 8 waves.  Block tile 64x256, wave tile 32x64:
// 8 WMMAs per K-step per wave, double-buffered LDS, 1 barrier per K-step.
// B tiles: async DMA global->LDS (no VGPR transit, no spills).
// Requires M%64==0, N%256==0, K%32==0.
// ---------------------------------------------------------------------------
__global__ __launch_bounds__(256)
__attribute__((amdgpu_waves_per_eu(1)))
void gemm_bf16_nt(
    const float* __restrict__ A, const unsigned short* __restrict__ B,
    float* __restrict__ C, int M, int N, int K)
{
  (void)M;
  __shared__ __align__(16) unsigned short a_s[2][64 * 40];   // 64x32 bf16
  __shared__ __align__(16) unsigned short b_s[2][256 * 40];  // 256x32 bf16

  const int tid  = threadIdx.x;
  const int lane = tid & 31;
  const int w    = tid >> 5;
  const int wm   = w & 1;          // 2 M subtiles of 32
  const int wn   = w >> 1;         // 4 N subtiles of 64
  const int half = lane >> 4;
  const int ln   = lane & 15;
  const int m0   = blockIdx.x << 6;
  const int n0   = blockIdx.y << 8;

  Acc acc[2][4];
  #pragma unroll
  for (int mf = 0; mf < 2; ++mf)
    #pragma unroll
    for (int nf = 0; nf < 4; ++nf)
      #pragma unroll
      for (int i = 0; i < 8; ++i) acc[mf][nf].f[i] = 0.0f;

  // staging geometry: A slot s in [0,512): row=s>>3, col4=(s&7)*4 (2/thread)
  //                   B row = tid (bf16), 4 x b128 per 32-elem K-chunk
  const int ar0 = (tid << 1) >> 3, ac0 = (((tid << 1) + 0) & 7) << 2;
  const int ar1 = ((tid << 1) + 1) >> 3, ac1 = (((tid << 1) + 1) & 7) << 2;
  const float* Abase = A + (size_t)m0 * K;
  const unsigned short* Brow = B + (size_t)(n0 + tid) * K;

  // prologue: stage K-chunk 0 into buffer 0
  float4 aF0 = *(const float4*)(Abase + (size_t)ar0 * K + ac0);
  float4 aF1 = *(const float4*)(Abase + (size_t)ar1 * K + ac1);
#if HAVE_ASYNC_LDS
  #pragma unroll
  for (int i = 0; i < 4; ++i)
    async_cp_b128(Brow + (i << 3), &b_s[0][tid * 40 + (i << 3)]);
#else
  U4 bU[4];
  #pragma unroll
  for (int i = 0; i < 4; ++i) bU[i] = *(const U4*)(Brow + (i << 3));
  #pragma unroll
  for (int i = 0; i < 4; ++i)
    *(U4*)&b_s[0][tid * 40 + (i << 3)] = bU[i];
#endif
  {
    U2 p;
    p.x = pk2bf(aF0.x, aF0.y); p.y = pk2bf(aF0.z, aF0.w);
    *(U2*)&a_s[0][ar0 * 40 + ac0] = p;
    p.x = pk2bf(aF1.x, aF1.y); p.y = pk2bf(aF1.z, aF1.w);
    *(U2*)&a_s[0][ar1 * 40 + ac1] = p;
  }

  int cur = 0;
  for (int kb = 0; kb < K; kb += 32) {
    wait_async0();                         // this wave's copies into buf[cur]
    __syncthreads();                       // buf[cur] ready; buf[cur^1] free
    const bool haveNext = (kb + 32) < K;

    if (haveNext) {
      // async DMA of next B chunk straight into LDS buf[cur^1]
#if HAVE_ASYNC_LDS
      #pragma unroll
      for (int i = 0; i < 4; ++i)
        async_cp_b128(Brow + kb + 32 + (i << 3),
                      &b_s[cur ^ 1][tid * 40 + (i << 3)]);
#else
      #pragma unroll
      for (int i = 0; i < 4; ++i)
        bU[i] = *(const U4*)(Brow + kb + 32 + (i << 3));
#endif
      // next A chunk via registers (needs f32->bf16 cvt)
      aF0 = *(const float4*)(Abase + (size_t)ar0 * K + kb + 32 + ac0);
      aF1 = *(const float4*)(Abase + (size_t)ar1 * K + kb + 32 + ac1);
    }
    if ((kb + 64) < K) {                   // L2 prefetch of chunk after next
      __builtin_prefetch(Abase + (size_t)ar0 * K + kb + 64 + ac0, 0, 3);
      __builtin_prefetch(Brow + kb + 64, 0, 3);
    }

    // compute from buf[cur]: 2 A frags x 4 B frags = 8 WMMAs
    Frag fa[2];
    #pragma unroll
    for (int mf = 0; mf < 2; ++mf) {
      const int r = (wm << 5) + (mf << 4) + ln;
      ld_half(fa[mf], 0, &a_s[cur][r * 40 + 8 * half]);
      ld_half(fa[mf], 4, &a_s[cur][r * 40 + 16 + 8 * half]);
    }
    #pragma unroll
    for (int nf = 0; nf < 4; ++nf) {
      Frag fb;
      const int br = (wn << 6) + (nf << 4) + ln;
      ld_half(fb, 0, &b_s[cur][br * 40 + 8 * half]);
      ld_half(fb, 4, &b_s[cur][br * 40 + 16 + 8 * half]);
      acc[0][nf].v = wmma_bf16(fa[0].v, fb.v, acc[0][nf].v);
      acc[1][nf].v = wmma_bf16(fa[1].v, fb.v, acc[1][nf].v);
    }

    if (haveNext) {                        // stage next A chunk into buf[cur^1]
      U2 p;
      p.x = pk2bf(aF0.x, aF0.y); p.y = pk2bf(aF0.z, aF0.w);
      *(U2*)&a_s[cur ^ 1][ar0 * 40 + ac0] = p;
      p.x = pk2bf(aF1.x, aF1.y); p.y = pk2bf(aF1.z, aF1.w);
      *(U2*)&a_s[cur ^ 1][ar1 * 40 + ac1] = p;
#if !HAVE_ASYNC_LDS
      #pragma unroll
      for (int i = 0; i < 4; ++i)
        *(U4*)&b_s[cur ^ 1][tid * 40 + (i << 3)] = bU[i];
#endif
    }
    cur ^= 1;
  }

  #pragma unroll
  for (int mf = 0; mf < 2; ++mf)
    #pragma unroll
    for (int i = 0; i < 8; ++i) {
      const size_t row = (size_t)(m0 + (wm << 5) + (mf << 4) + i + 8 * half);
      #pragma unroll
      for (int nf = 0; nf < 4; ++nf)
        C[row * N + n0 + (wn << 6) + (nf << 4) + ln] = acc[mf][nf].f[i];
    }
}

// ---------------------------------------------------------------------------
// Interleaved-pair RoPE, in place on (S, H, 128) fp32.
// ---------------------------------------------------------------------------
__global__ __launch_bounds__(256) void rope_kernel(
    float* __restrict__ X, int nheads, int total_pairs)
{
  const int idx = blockIdx.x * 256 + threadIdx.x;
  if (idx >= total_pairs) return;
  const int j   = idx & 63;
  const int rem = idx >> 6;
  const int h   = rem % nheads;
  const int s   = rem / nheads;
  const float inv = __powf(10000.0f, -(float)(2 * j) * (1.0f / 128.0f));
  float sn, cs;
  __sincosf((float)s * inv, &sn, &cs);
  const size_t base = ((size_t)s * nheads + h) * 128 + (j << 1);
  const float x1 = X[base], x2 = X[base + 1];
  X[base]     = x1 * cs - x2 * sn;
  X[base + 1] = x1 * sn + x2 * cs;
}

// ---------------------------------------------------------------------------
// Flash attention (non-causal, full 2048 keys), GQA 32q/8kv heads, hd=128.
// Block = 8 waves on the same head; K/V tiles staged cooperatively; each wave
// owns 16 query rows; online softmax in registers; 16 WMMAs per 32-key step.
// Grid: (16, 32).
// ---------------------------------------------------------------------------
__global__ __launch_bounds__(256)
__attribute__((amdgpu_waves_per_eu(1)))
void attn_kernel(
    const float* __restrict__ Q, const float* __restrict__ Kc,
    const float* __restrict__ Vc, float* __restrict__ O)
{
  __shared__ __align__(16) unsigned short kbuf[32 * 136];      // [key][d] bf16
  __shared__ __align__(16) unsigned short vbt [128 * 40];      // [d][key] bf16
  __shared__ __align__(16) unsigned short pbuf[8 * 16 * 40];   // per-wave P

  const int tid  = threadIdx.x;
  const int lane = tid & 31;
  const int w    = tid >> 5;
  const int half = lane >> 4;
  const int ln   = lane & 15;
  const int h    = blockIdx.y;
  const int kvh  = h >> 2;
  const int q0   = (blockIdx.x << 7) + (w << 4);

  // Q A-fragments (4 x 16x32 over hd=128), loaded once, fp32->bf16 in regs
  Frag qa[4];
  const float* qrow = Q + (size_t)(q0 + ln) * 4096 + h * 128;
  #pragma unroll
  for (int c = 0; c < 4; ++c) {
    const int d0 = c << 5;
    const float4 f0 = *(const float4*)(qrow + d0 + 8 * half);
    const float4 f1 = *(const float4*)(qrow + d0 + 8 * half + 4);
    const float4 f2 = *(const float4*)(qrow + d0 + 16 + 8 * half);
    const float4 f3 = *(const float4*)(qrow + d0 + 16 + 8 * half + 4);
    qa[c].u[0] = pk2bf(f0.x, f0.y); qa[c].u[1] = pk2bf(f0.z, f0.w);
    qa[c].u[2] = pk2bf(f1.x, f1.y); qa[c].u[3] = pk2bf(f1.z, f1.w);
    qa[c].u[4] = pk2bf(f2.x, f2.y); qa[c].u[5] = pk2bf(f2.z, f2.w);
    qa[c].u[6] = pk2bf(f3.x, f3.y); qa[c].u[7] = pk2bf(f3.z, f3.w);
  }

  float runmax[8], runsum[8];
  Acc o[8];
  #pragma unroll
  for (int i = 0; i < 8; ++i) { runmax[i] = -1e30f; runsum[i] = 0.0f; }
  #pragma unroll
  for (int f = 0; f < 8; ++f)
    #pragma unroll
    for (int i = 0; i < 8; ++i) o[f].f[i] = 0.0f;

  const float scale = 0.08838834764831845f;  // 1/sqrt(128)
  unsigned short* pw = &pbuf[w * (16 * 40)];

  for (int kb = 0; kb < 2048; kb += 32) {
    __syncthreads();
    #pragma unroll
    for (int i = 0; i < 4; ++i) {   // stage 32 keys x 128d of K and V
      const int s   = (tid << 2) + i;
      const int key = s >> 5;
      const int d   = (s & 31) << 2;
      const size_t src = (size_t)(kb + key) * 1024 + kvh * 128 + d;
      const float4 kf = *(const float4*)(Kc + src);
      U2 pk; pk.x = pk2bf(kf.x, kf.y); pk.y = pk2bf(kf.z, kf.w);
      *(U2*)&kbuf[key * 136 + d] = pk;
      const float4 vf = *(const float4*)(Vc + src);
      vbt[(d + 0) * 40 + key] = f2bf(vf.x);
      vbt[(d + 1) * 40 + key] = f2bf(vf.y);
      vbt[(d + 2) * 40 + key] = f2bf(vf.z);
      vbt[(d + 3) * 40 + key] = f2bf(vf.w);
      if (i == 0 && kb + 32 < 2048) {     // L2 prefetch of next tile
        __builtin_prefetch(Kc + src + 32 * 1024, 0, 3);
        __builtin_prefetch(Vc + src + 32 * 1024, 0, 3);
      }
    }
    __syncthreads();

    // scores: two 16x16 tiles (keys 0-15, 16-31), 4 chained WMMA each
    Acc c0, c1;
    #pragma unroll
    for (int i = 0; i < 8; ++i) { c0.f[i] = 0.0f; c1.f[i] = 0.0f; }
    #pragma unroll
    for (int c = 0; c < 4; ++c) {
      const int d0 = c << 5;
      Frag kb0, kb1;
      ld_half(kb0, 0, &kbuf[ln * 136 + d0 + 8 * half]);
      ld_half(kb0, 4, &kbuf[ln * 136 + d0 + 16 + 8 * half]);
      ld_half(kb1, 0, &kbuf[(16 + ln) * 136 + d0 + 8 * half]);
      ld_half(kb1, 4, &kbuf[(16 + ln) * 136 + d0 + 16 + 8 * half]);
      c0.v = wmma_bf16(qa[c].v, kb0.v, c0.v);
      c1.v = wmma_bf16(qa[c].v, kb1.v, c1.v);
    }

    // online softmax: rows 0-7 in lanes 0-15, rows 8-15 in lanes 16-31
    float p0[8], p1[8], fac[8];
    #pragma unroll
    for (int i = 0; i < 8; ++i) {
      const float s0 = c0.f[i] * scale;
      const float s1 = c1.f[i] * scale;
      float tm = fmaxf(s0, s1);
      #pragma unroll
      for (int m = 1; m < 16; m <<= 1) tm = fmaxf(tm, __shfl_xor(tm, m, 32));
      const float nm = fmaxf(runmax[i], tm);
      fac[i] = __expf(runmax[i] - nm);
      p0[i]  = __expf(s0 - nm);
      p1[i]  = __expf(s1 - nm);
      float ts = p0[i] + p1[i];
      #pragma unroll
      for (int m = 1; m < 16; m <<= 1) ts += __shfl_xor(ts, m, 32);
      runsum[i] = runsum[i] * fac[i] + ts;
      runmax[i] = nm;
    }
    #pragma unroll
    for (int f = 0; f < 8; ++f)
      #pragma unroll
      for (int i = 0; i < 8; ++i) o[f].f[i] *= fac[i];

    // restage P (16x32 bf16) into A-fragment layout via per-wave LDS pad
    #pragma unroll
    for (int i = 0; i < 8; ++i) {
      const int r = i + 8 * half;
      pw[r * 40 + ln]      = f2bf(p0[i]);
      pw[r * 40 + 16 + ln] = f2bf(p1[i]);
    }
    Frag pa;
    ld_half(pa, 0, &pw[ln * 40 + 8 * half]);
    ld_half(pa, 4, &pw[ln * 40 + 16 + 8 * half]);

    // O += P * V   (8 hd-column blocks of 16)
    #pragma unroll
    for (int f = 0; f < 8; ++f) {
      Frag vb;
      const int vr = (f << 4) + ln;
      ld_half(vb, 0, &vbt[vr * 40 + 8 * half]);
      ld_half(vb, 4, &vbt[vr * 40 + 16 + 8 * half]);
      o[f].v = wmma_bf16(pa.v, vb.v, o[f].v);
    }
  }

  #pragma unroll
  for (int i = 0; i < 8; ++i) {
    const float inv = 1.0f / runsum[i];
    const size_t row = (size_t)(q0 + i + 8 * half) * 4096 + h * 128;
    #pragma unroll
    for (int f = 0; f < 8; ++f)
      O[row + (f << 4) + ln] = o[f].f[i] * inv;
  }
}

// ---------------------------------------------------------------------------
extern "C" void kernel_launch(void* const* d_in, const int* in_sizes, int n_in,
                              void* d_out, int out_size, void* d_ws, size_t ws_size,
                              hipStream_t stream) {
  (void)in_sizes; (void)n_in; (void)out_size; (void)ws_size;
  const float* x  = (const float*)d_in[0];
  const float* wq = (const float*)d_in[1];
  const float* wk = (const float*)d_in[2];
  const float* wv = (const float*)d_in[3];
  const float* wo = (const float*)d_in[4];
  // d_in[5] = start_pos (0 in this workload; prefill path assumed)

  float* Qw = (float*)d_ws;                       // 2048*4096 f32
  float* Kw = Qw + (size_t)2048 * 4096;           // 2048*1024 f32
  float* Vw = Kw + (size_t)2048 * 1024;           // 2048*1024 f32
  float* Aw = Vw + (size_t)2048 * 1024;           // 2048*4096 f32
  unsigned short* wqb = (unsigned short*)(Aw + (size_t)2048 * 4096);
  unsigned short* wkb = wqb + (size_t)4096 * 4096;
  unsigned short* wvb = wkb + (size_t)1024 * 4096;
  unsigned short* wob = wvb + (size_t)1024 * 4096;
  float* out = (float*)d_out;

  const dim3 blk(256);
  const int n8q = (4096 * 4096) / 8, n8k = (1024 * 4096) / 8;
  cvt_f32_bf16<<<n8q / 256, blk, 0, stream>>>(wq, wqb, n8q);
  cvt_f32_bf16<<<n8k / 256, blk, 0, stream>>>(wk, wkb, n8k);
  cvt_f32_bf16<<<n8k / 256, blk, 0, stream>>>(wv, wvb, n8k);
  cvt_f32_bf16<<<n8q / 256, blk, 0, stream>>>(wo, wob, n8q);

  gemm_bf16_nt<<<dim3(32, 16), blk, 0, stream>>>(x,  wqb, Qw, 2048, 4096, 4096);
  gemm_bf16_nt<<<dim3(32,  4), blk, 0, stream>>>(x,  wkb, Kw, 2048, 1024, 4096);
  gemm_bf16_nt<<<dim3(32,  4), blk, 0, stream>>>(x,  wvb, Vw, 2048, 1024, 4096);

  rope_kernel<<<(2048 * 32 * 64 + 255) / 256, blk, 0, stream>>>(Qw, 32, 2048 * 32 * 64);
  rope_kernel<<<(2048 *  8 * 64 + 255) / 256, blk, 0, stream>>>(Kw,  8, 2048 *  8 * 64);

  attn_kernel<<<dim3(16, 32), blk, 0, stream>>>(Qw, Kw, Vw, Aw);

  gemm_bf16_nt<<<dim3(32, 16), blk, 0, stream>>>(Aw, wob, out, 2048, 4096, 4096);
}
